// TokenChoiceMoRBlock_81784767251165
// MI455X (gfx1250) — compile-verified, matrix operand order checked by reading, not run on hardware
//
#include <hip/hip_runtime.h>
#include <hip/hip_bf16.h>
#include <cstdint>

// Problem dims (fixed by reference setup_inputs)
#define NT    8192     // B*S tokens
#define DDIM  2048
#define FDIM  8192
#define LVLS  2
#define EPSF  1e-6f
#define ALPHAF 1.0f

// GEMM tiling
#define BM 64
#define BN 128
#define BK 64
#define NB16 (BN / 16)                              // 16-col fragment groups per tile
#define BFRAG_ELEMS ((BK / 32) * NB16 * 32 * 16)    // packed B tile elements (8192)

typedef __attribute__((ext_vector_type(16))) __bf16 v16bf;
typedef __attribute__((ext_vector_type(8)))  float  v8f;
typedef unsigned __attribute__((ext_vector_type(4))) u32x4;
typedef unsigned __attribute__((ext_vector_type(8))) u32x8;

__device__ __forceinline__ __bf16 f2bf(float f) {
  // round-to-nearest-even fp32 -> bf16
  unsigned u = __builtin_bit_cast(unsigned, f);
  unsigned r = (u + 0x7FFFu + ((u >> 16) & 1u)) >> 16;
  unsigned short s = (unsigned short)r;
  return __builtin_bit_cast(__bf16, s);
}

__device__ __forceinline__ v8f wmma_bf16(v16bf a, v16bf b, v8f c) {
  // (neg_a, A, neg_b, B, c_mod, C, reuse_a, reuse_b)
  return __builtin_amdgcn_wmma_f32_16x16x32_bf16(false, a, false, b, (short)0, c,
                                                 false, false);
}

union frag_u {
  uint4 q[2];
  v16bf v;
};

// A fragment: 16x32 bf16 from row-major LDS tile. Lane (half*16+m) needs K runs
// [half*8, half*8+8) and [16+half*8, ...+8) of row m: two ds_load_b128.
__device__ __forceinline__ v16bf load_frag_a(const __bf16* smem, int lda, int m0, int kk) {
  const int lane = threadIdx.x & 31;
  const int half = lane >> 4, m = lane & 15;
  const __bf16* p = smem + (size_t)(m0 + m) * lda + kk + half * 8;
  frag_u u;
  u.q[0] = *(const uint4*)p;
  u.q[1] = *(const uint4*)(p + 16);
  return u.v;
}

// B fragment from fragment-packed LDS: [kb32][nb][lane][16] -> one lane's 16
// bf16 are contiguous (32B) -> two ds_load_b128.
__device__ __forceinline__ v16bf load_frag_b(const __bf16* sB, int kb32, int nb) {
  const int lane = threadIdx.x & 31;
  const __bf16* p = sB + ((size_t)((kb32 * NB16 + nb) * 32 + lane)) * 16;
  frag_u u;
  u.q[0] = *(const uint4*)(p);
  u.q[1] = *(const uint4*)(p + 8);
  return u.v;
}

// TDM DMA of one contiguous pre-packed 16KB B tile: global -> LDS.
// D# per ISA ch.8: group0 = {count=1, lds_addr, global_addr, type=2};
// group1 = {mask=0, data_size=1(2B), tensor_dim0=tile_dim0=BFRAG_ELEMS,
//           tensor_dim1=tile_dim1=1, dim0_stride=BFRAG_ELEMS}.
// VADDR2/VADDR3 omitted (NULL) -> 2-group form (<=2D tensor).
// Low 32 bits of a generic LDS pointer are the LDS byte offset (ISA 10.2).
__device__ __forceinline__ void tdm_load_packed_tile(const __bf16* gsrc, __bf16* ldst) {
  const unsigned long long ga = (unsigned long long)(const void*)gsrc;
  const unsigned lds = (unsigned)(unsigned long long)(void*)ldst;
  u32x4 g0;
  g0.x = 1u;                                                   // count=1 (valid user D#)
  g0.y = lds;                                                  // lds_addr
  g0.z = (unsigned)ga;                                         // global_addr[31:0]
  g0.w = (unsigned)((ga >> 32) & 0x01FFFFFFull) | (2u << 30);  // addr[56:32] | type=2
  u32x8 g1;
  g1[0] = 1u << 16;                                            // data_size=1 (2 bytes)
  g1[1] = ((unsigned)BFRAG_ELEMS & 0xFFFFu) << 16;             // tensor_dim0[15:0]
  g1[2] = ((unsigned)BFRAG_ELEMS >> 16) | (1u << 16);          // dim0[31:16], tensor_dim1=1
  g1[3] = ((unsigned)BFRAG_ELEMS & 0xFFFFu) << 16;             // tile_dim0
  g1[4] = 1u;                                                  // tile_dim1=1, tile_dim2=0
  g1[5] = (unsigned)BFRAG_ELEMS;                               // tensor_dim0_stride[31:0]
  g1[6] = 0u;
  g1[7] = 0u;
  asm volatile("tensor_load_to_lds %0, %1" :: "s"(g0), "s"(g1) : "memory");
}

// ---------------------------------------------------------------- kernels ---

__global__ void init_kernel(int* cnt) {
  cnt[0] = 0;
  cnt[1] = 0;
}

// One-shot fp32 -> fragment-packed-bf16 weight conversion (per level, per tile).
// Coalesced b32 row reads; one 16B packed store per group per thread.
__global__ __launch_bounds__(256) void pack_kernel(const float* __restrict__ W,
                                                   __bf16* __restrict__ P,
                                                   int K, int N) {
  const int lvl = blockIdx.z;
  const int kc = blockIdx.y;                       // K / BK
  const int nt = blockIdx.x;                       // N / BN
  const float* base = W + (size_t)lvl * K * N + (size_t)kc * BK * N + (size_t)nt * BN;
  __bf16* out = P + (size_t)lvl * K * N + ((size_t)kc * (N / BN) + nt) * BFRAG_ELEMS;
#pragma unroll
  for (int i = 0; i < (BK * BN) / (256 * 8); ++i) {
    const int g = threadIdx.x + i * 256;
    const int n = g % BN;
    const int r = g / BN;                          // 0 .. (BK/8 - 1)
    const int half = r & 1;
    const int grp = (r >> 1) & 1;
    const int kb32 = r >> 2;
    const int krow = kb32 * 32 + grp * 16 + half * 8;
    const float* src = base + (size_t)krow * N + n;
    __bf16 tmp[8];
#pragma unroll
    for (int j = 0; j < 8; ++j) tmp[j] = f2bf(src[(size_t)j * N]);
    __bf16* dst = out + ((size_t)((kb32 * NB16 + (n >> 4)) * 32 + half * 16 + (n & 15))) * 16 +
                  grp * 8;
    *(uint4*)dst = *(const uint4*)tmp;
  }
}

// One wave32 per token: router logits, softmax/top-1, z-loss term, RMSNorm,
// bf16 normalized-token write, compaction into per-level lists.
__global__ __launch_bounds__(256) void router_kernel(
    const float* __restrict__ X, const float* __restrict__ RW,
    const float* __restrict__ NW, int* __restrict__ cnt, int* __restrict__ list,
    float* __restrict__ wtok, float* __restrict__ zz, __bf16* __restrict__ Xn) {
  const int wave = threadIdx.x >> 5;
  const int lane = threadIdx.x & 31;
  const int t = blockIdx.x * 8 + wave;
  if (t >= NT) return;
  const float* x = X + (size_t)t * DDIM;

  float s0 = 0.f, s1 = 0.f, ss = 0.f;
  for (int k = lane; k < DDIM; k += 32) {
    const float v = x[k];
    s0 += v * RW[k];
    s1 += v * RW[DDIM + k];
    ss += v * v;
  }
#pragma unroll
  for (int off = 16; off; off >>= 1) {
    s0 += __shfl_xor(s0, off, 32);
    s1 += __shfl_xor(s1, off, 32);
    ss += __shfl_xor(ss, off, 32);
  }

  const float m = fmaxf(s0, s1);
  const float e0 = __expf(s0 - m), e1 = __expf(s1 - m);
  const float denom = e0 + e1;
  const int idx = (s0 >= s1) ? 0 : 1;           // argmax, first-on-tie
  const float wsel = ((idx == 0) ? e0 : e1) / denom * ALPHAF;
  const float z = m + __logf(denom);            // logsumexp

  if (lane == 0) {
    const int pos = atomicAdd(&cnt[idx], 1);
    list[idx * NT + pos] = t;
    wtok[t] = wsel;
    zz[t] = z * z;
  }

  const float rinv = rsqrtf(ss / (float)DDIM + EPSF);
  const float* nw = NW + (size_t)idx * DDIM;
  for (int k = lane; k < DDIM; k += 32)
    Xn[(size_t)t * DDIM + k] = f2bf(x[k] * rinv * nw[k]);
}

// Deterministic fixed-order reduction of z^2 terms.
__global__ __launch_bounds__(256) void zloss_kernel(const float* __restrict__ zz,
                                                    float* __restrict__ out_scalar) {
  __shared__ float red[256];
  float s = 0.f;
  for (int i = threadIdx.x; i < NT; i += 256) s += zz[i];
  red[threadIdx.x] = s;
  __syncthreads();
  for (int o = 128; o; o >>= 1) {
    if (threadIdx.x < (unsigned)o) red[threadIdx.x] += red[threadIdx.x + o];
    __syncthreads();
  }
  if (threadIdx.x == 0) *out_scalar = red[0] / (float)NT;
}

// H[tok, f] = silu(Xn[tok,:] @ Wg[:, f]) * (Xn[tok,:] @ Wu[:, f]), grouped by level.
// Wave grid 2(M) x 4(N): each wave owns 32x32, i.e. 2x2 fragment tiles/matrix.
__global__ __launch_bounds__(256) void mlp_up_kernel(
    const __bf16* __restrict__ Xn, const __bf16* __restrict__ Wgp,
    const __bf16* __restrict__ Wup, const int* __restrict__ list,
    const int* __restrict__ cnt, __bf16* __restrict__ H) {
  const int lvl = blockIdx.z;
  const int n_tok = cnt[lvl];
  const int m0 = blockIdx.y * BM;
  if (m0 >= n_tok) return;
  const int f0 = blockIdx.x * BN;

  __shared__ __bf16 sA[BM][BK + 16];       // 160B row stride, 16B aligned
  __shared__ __bf16 sG[BFRAG_ELEMS];       // fragment-packed (TDM destination)
  __shared__ __bf16 sU[BFRAG_ELEMS];
  __shared__ int sTok[BM];

  const int tid = threadIdx.x;
  const int wave = tid >> 5;
  const int wm = wave & 1;   // 32-row half
  const int wn = wave >> 1;  // 32-col quarter

  if (tid < BM) {
    const int row = m0 + tid;
    sTok[tid] = (row < n_tok) ? list[lvl * NT + row] : -1;
  }
  __syncthreads();

  const __bf16* gbase = Wgp + (size_t)lvl * DDIM * FDIM;
  const __bf16* ubase = Wup + (size_t)lvl * DDIM * FDIM;

  v8f cg[4] = {}, cu[4] = {};

  for (int kk = 0; kk < DDIM; kk += BK) {
    const size_t tile = ((size_t)(kk / BK) * (FDIM / BN) + (f0 / BN)) * BFRAG_ELEMS;
    // TDM DMA both packed B tiles (wave 0 issues; DMA runs async to the block)
    if (wave == 0) {
      tdm_load_packed_tile(gbase + tile, sG);
      tdm_load_packed_tile(ubase + tile, sU);
    }
    // stage A (gathered token rows, already bf16): 32B per thread
    {
      const int row = tid >> 2;
      const int col = (tid & 3) * 16;
      const int tok = sTok[row];
      const __bf16* src = Xn + (size_t)(tok < 0 ? 0 : tok) * DDIM + kk + col;
      const uint4 va = *(const uint4*)src;
      const uint4 vb = *(const uint4*)(src + 8);
      *(uint4*)&sA[row][col] = va;
      *(uint4*)&sA[row][col + 8] = vb;
    }
    if (kk + BK < DDIM) {   // prefetch next K-chunk tiles (global_prefetch_b8)
      const size_t next = tile + (size_t)(FDIM / BN) * BFRAG_ELEMS;
      __builtin_prefetch(gbase + next, 0, 1);
      __builtin_prefetch(ubase + next, 0, 1);
    }
    if (wave == 0) __builtin_amdgcn_s_wait_tensorcnt(0);
    __syncthreads();

#pragma unroll
    for (int ks = 0; ks < BK; ks += 32) {
      const int kb32 = ks >> 5;
      const v16bf a0 = load_frag_a(&sA[0][0], BK + 16, wm * 32, ks);
      const v16bf a1 = load_frag_a(&sA[0][0], BK + 16, wm * 32 + 16, ks);
      const v16bf bg0 = load_frag_b(sG, kb32, wn * 2);
      const v16bf bg1 = load_frag_b(sG, kb32, wn * 2 + 1);
      const v16bf bu0 = load_frag_b(sU, kb32, wn * 2);
      const v16bf bu1 = load_frag_b(sU, kb32, wn * 2 + 1);
      cg[0] = wmma_bf16(a0, bg0, cg[0]);
      cg[1] = wmma_bf16(a0, bg1, cg[1]);
      cg[2] = wmma_bf16(a1, bg0, cg[2]);
      cg[3] = wmma_bf16(a1, bg1, cg[3]);
      cu[0] = wmma_bf16(a0, bu0, cu[0]);
      cu[1] = wmma_bf16(a0, bu1, cu[1]);
      cu[2] = wmma_bf16(a1, bu0, cu[2]);
      cu[3] = wmma_bf16(a1, bu1, cu[3]);
    }
    __syncthreads();
  }

  // epilogue: silu(g)*u -> bf16 H, scattered by token id.
  // C layout: VGPR r holds (M = r + 8*(lane>=16), N = lane%16).
  const int lane = tid & 31;
  const int half = lane >> 4, n = lane & 15;
#pragma unroll
  for (int mi = 0; mi < 2; ++mi) {
#pragma unroll
    for (int ni = 0; ni < 2; ++ni) {
      const v8f g = cg[mi * 2 + ni];
      const v8f u = cu[mi * 2 + ni];
#pragma unroll
      for (int r = 0; r < 8; ++r) {
        const int mrow = wm * 32 + mi * 16 + half * 8 + r;
        const int tok = sTok[mrow];
        if (tok >= 0) {
          const float gv = g[r];
          const float hv = (gv / (1.f + __expf(-gv))) * u[r];
          H[(size_t)tok * FDIM + f0 + wn * 32 + ni * 16 + n] = f2bf(hv);
        }
      }
    }
  }
}

// out[tok, d] = x[tok,d]*(1+w) + w * (H[tok,:] @ Wd[:, d]), grouped by level.
__global__ __launch_bounds__(256) void mlp_down_kernel(
    const __bf16* __restrict__ H, const __bf16* __restrict__ Wdp,
    const float* __restrict__ X, const float* __restrict__ wtok,
    const int* __restrict__ list, const int* __restrict__ cnt,
    float* __restrict__ out) {
  const int lvl = blockIdx.z;
  const int n_tok = cnt[lvl];
  const int m0 = blockIdx.y * BM;
  if (m0 >= n_tok) return;
  const int d0 = blockIdx.x * BN;

  __shared__ __bf16 sH[BM][BK + 16];
  __shared__ __bf16 sW[BFRAG_ELEMS];
  __shared__ int sTok[BM];
  __shared__ float sWt[BM];

  const int tid = threadIdx.x;
  const int wave = tid >> 5;
  const int wm = wave & 1;
  const int wn = wave >> 1;

  if (tid < BM) {
    const int row = m0 + tid;
    const int tok = (row < n_tok) ? list[lvl * NT + row] : -1;
    sTok[tid] = tok;
    sWt[tid] = (tok >= 0) ? wtok[tok] : 0.f;
  }
  __syncthreads();

  const __bf16* wbase = Wdp + (size_t)lvl * FDIM * DDIM;

  v8f c[4] = {};

  for (int kk = 0; kk < FDIM; kk += BK) {
    const size_t tile = ((size_t)(kk / BK) * (DDIM / BN) + (d0 / BN)) * BFRAG_ELEMS;
    if (wave == 0) {
      tdm_load_packed_tile(wbase + tile, sW);
    }
    {
      const int row = tid >> 2;
      const int col = (tid & 3) * 16;
      const int tok = sTok[row];
      const __bf16* src = H + (size_t)(tok < 0 ? 0 : tok) * FDIM + kk + col;
      const uint4 va = *(const uint4*)src;
      const uint4 vb = *(const uint4*)(src + 8);
      *(uint4*)&sH[row][col] = va;
      *(uint4*)&sH[row][col + 8] = vb;
    }
    if (kk + BK < FDIM) {
      __builtin_prefetch(wbase + tile + (size_t)(DDIM / BN) * BFRAG_ELEMS, 0, 1);
    }
    if (wave == 0) __builtin_amdgcn_s_wait_tensorcnt(0);
    __syncthreads();

#pragma unroll
    for (int ks = 0; ks < BK; ks += 32) {
      const int kb32 = ks >> 5;
      const v16bf a0 = load_frag_a(&sH[0][0], BK + 16, wm * 32, ks);
      const v16bf a1 = load_frag_a(&sH[0][0], BK + 16, wm * 32 + 16, ks);
      const v16bf b0 = load_frag_b(sW, kb32, wn * 2);
      const v16bf b1 = load_frag_b(sW, kb32, wn * 2 + 1);
      c[0] = wmma_bf16(a0, b0, c[0]);
      c[1] = wmma_bf16(a0, b1, c[1]);
      c[2] = wmma_bf16(a1, b0, c[2]);
      c[3] = wmma_bf16(a1, b1, c[3]);
    }
    __syncthreads();
  }

  const int lane = tid & 31;
  const int half = lane >> 4, n = lane & 15;
#pragma unroll
  for (int mi = 0; mi < 2; ++mi) {
#pragma unroll
    for (int ni = 0; ni < 2; ++ni) {
      const v8f acc = c[mi * 2 + ni];
#pragma unroll
      for (int r = 0; r < 8; ++r) {
        const int mrow = wm * 32 + mi * 16 + half * 8 + r;
        const int tok = sTok[mrow];
        if (tok >= 0) {
          const float w = sWt[mrow];
          const int col = d0 + wn * 32 + ni * 16 + n;
          out[(size_t)tok * DDIM + col] =
              X[(size_t)tok * DDIM + col] * (1.f + w) + w * acc[r];
        }
      }
    }
  }
}

// ------------------------------------------------------------------ launch --

extern "C" void kernel_launch(void* const* d_in, const int* in_sizes, int n_in,
                              void* d_out, int out_size, void* d_ws, size_t ws_size,
                              hipStream_t stream) {
  const float* X  = (const float*)d_in[0];   // (B,S,D)
  const float* RW = (const float*)d_in[1];   // (L,D)
  const float* Wg = (const float*)d_in[2];   // (L,D,F)
  const float* Wu = (const float*)d_in[3];   // (L,D,F)
  const float* Wd = (const float*)d_in[4];   // (L,F,D)
  const float* NW = (const float*)d_in[5];   // (L,D)
  float* out = (float*)d_out;                // (B,S,D) flat + 1 scalar (z-loss)

  char* ws = (char*)d_ws;
  int* cnt    = (int*)ws;                                   // 2 ints
  float* wtok = (float*)(ws + 256);                         // NT floats
  float* zz   = (float*)(ws + 256 + (size_t)NT * 4);        // NT floats
  int* list   = (int*)(ws + 256 + (size_t)NT * 8);          // LVLS*NT ints
  size_t off = 256 + (size_t)NT * 8 + (size_t)LVLS * NT * 4;
  off = (off + 255) & ~(size_t)255;
  __bf16* Xn  = (__bf16*)(ws + off); off += (size_t)NT * DDIM * 2;          // 32 MB
  __bf16* H   = (__bf16*)(ws + off); off += (size_t)NT * FDIM * 2;          // 128 MB
  __bf16* Wgp = (__bf16*)(ws + off); off += (size_t)LVLS * DDIM * FDIM * 2; // 67 MB
  __bf16* Wup = (__bf16*)(ws + off); off += (size_t)LVLS * DDIM * FDIM * 2; // 67 MB
  __bf16* Wdp = (__bf16*)(ws + off);                                         // 67 MB

  init_kernel<<<1, 1, 0, stream>>>(cnt);
  router_kernel<<<NT / 8, 256, 0, stream>>>(X, RW, NW, cnt, list, wtok, zz, Xn);
  zloss_kernel<<<1, 256, 0, stream>>>(zz, out + (size_t)NT * DDIM);
  pack_kernel<<<dim3(FDIM / BN, DDIM / BK, LVLS), 256, 0, stream>>>(Wg, Wgp, DDIM, FDIM);
  pack_kernel<<<dim3(FDIM / BN, DDIM / BK, LVLS), 256, 0, stream>>>(Wu, Wup, DDIM, FDIM);
  pack_kernel<<<dim3(DDIM / BN, FDIM / BK, LVLS), 256, 0, stream>>>(Wd, Wdp, FDIM, DDIM);
  mlp_up_kernel<<<dim3(FDIM / BN, NT / BM, LVLS), 256, 0, stream>>>(
      Xn, Wgp, Wup, list, cnt, H);
  mlp_down_kernel<<<dim3(DDIM / BN, NT / BM, LVLS), 256, 0, stream>>>(
      H, Wdp, X, wtok, list, cnt, out);
}
